// NeuralSheet_7791070675928
// MI455X (gfx1250) — matrix operand order repaired and checked by reading, not compiled, not run on hardware
//
#include <hip/hip_runtime.h>
#include <hip/hip_bf16.h>

typedef __attribute__((ext_vector_type(16))) _Float16 v16h;
typedef __attribute__((ext_vector_type(8)))  float    v8f;

#define NSHEET   64
#define N2       4096
#define WWIN     21
#define PADW     10
#define KTOT     441          // 21*21
#define NCH      14           // ceil(441/32) K-chunks of 32
#define RF2      169          // 13*13
#define MAXITERS 100
#define TOLV     0.003f
#define L4STR    2.0f
#define NWG      64
#define NTHR     128          // 4 wave32 waves
#define PATCHW   84           // 64 + 2*PADW

// -------- workspace layout (float units) ----------------------------------
//  [0      .. 4096)   net_afferent
//  [4096   .. 8192)   mid-normalization sums
//  [8192   .. 12288)  resp buffer 0
//  [12288  .. 16384)  resp buffer 1
//  at float offset 16384, as unsigned: [0]=count [1]=gen [2]=changeBits [3]=done
#define WS_NETAFF 0
#define WS_SUMS   4096
#define WS_RESP0  8192
#define WS_RESP1  12288
#define WS_CTL    16384

// ---------------------------------------------------------------------------
__global__ void k_init(float* ws) {
  int i = blockIdx.x * blockDim.x + threadIdx.x;
  if (i < N2) ws[WS_RESP0 + i] = 0.0f;      // resp starts at zero
  if (i == 0) {
    unsigned* ctl = (unsigned*)(ws + WS_CTL);
    ctl[0] = 0u; ctl[1] = 0u; ctl[2] = 0u; ctl[3] = 0u;
  }
}

// ---------------------------------------------------------------------------
// net_afferent[n] = sum(tiles * aff_w_normalized) - threshold   (AFF_STRENGTH=1)
__global__ void k_afferent(const float* __restrict__ input,
                           const float* __restrict__ aff,
                           const float* __restrict__ bias,
                           const float* __restrict__ thr,
                           const int*   __restrict__ rf,
                           float* ws) {
  int n = blockIdx.x * blockDim.x + threadIdx.x;
  if (n >= N2) return;
  const float* a = aff  + (size_t)n * RF2;
  const float* b = bias + (size_t)n * RF2;
  const int*   g = rf   + (size_t)n * RF2;
  float s1 = 0.f, s2 = 0.f;
  for (int k = 0; k < RF2; ++k) {
    float w = a[k] * b[k];
    s1 += w;
    s2 += w * input[g[k]];
  }
  ws[WS_NETAFF + n] = s2 / (s1 + 1e-11f) - thr[n];
}

// ---------------------------------------------------------------------------
// sums[n] = sum over 64x64 of l4_correlations*(1-masks)   (reads 128MB once)
__global__ void k_midsum(const float* __restrict__ l4c,
                         const float* __restrict__ mask,
                         float* ws) {
  int n = blockIdx.x;
  const float* c = l4c  + (size_t)n * N2;
  const float* m = mask + (size_t)n * N2;
  float p = 0.f;
  for (int k = threadIdx.x; k < N2; k += blockDim.x)
    p += c[k] * (1.0f - m[k]);
  __shared__ float red[256];
  red[threadIdx.x] = p;
  __syncthreads();
  for (int s = 128; s > 0; s >>= 1) {
    if (threadIdx.x < s) red[threadIdx.x] += red[threadIdx.x + s];
    __syncthreads();
  }
  if (threadIdx.x == 0) ws[WS_SUMS + n] = red[0];
}

// ---------------------------------------------------------------------------
// Persistent lock-step iteration kernel. One block per sheet row, one wave
// per 16-neuron column group. Crop weights are built once directly in the
// WMMA B-operand register layout (f16) and stay register-resident for all
// iterations; per-iteration work is LDS patch gathers + 14 WMMAs + diag.
__global__ void __launch_bounds__(NTHR, 1)
k_iterate(const float* __restrict__ wexc,
          const float* __restrict__ l4c,
          const float* __restrict__ mask,
          float* ws, float* out) {
  const int tid  = threadIdx.x;
  const int lane = tid & 31;
  const int w    = tid >> 5;          // wave 0..3
  const int r    = blockIdx.x;        // sheet row
  const int c0   = w * 16;
  const int col  = c0 + (lane & 15);  // this lane's neuron column (M for A, N for B)
  const int n    = r * NSHEET + col;

  float*    netaff = ws + WS_NETAFF;
  float*    sums   = ws + WS_SUMS;
  float*    respb[2] = { ws + WS_RESP0, ws + WS_RESP1 };
  unsigned* ctl    = (unsigned*)(ws + WS_CTL);

  __shared__ _Float16 patch[WWIN * PATCHW];   // padded resp window rows r..r+20
  __shared__ float    diag[4][16];
  __shared__ int      sdone;

  // ---- build B operand (crops) in registers, contiguous K order: K = s + (lane&16)
  v16h b[NCH];
  {
    float inv = 1.0f / (sums[n] + 1e-11f);
    const int khi = lane & 16;        // lanes 16..31 hold K+16
#pragma unroll
    for (int q = 0; q < NCH; ++q) {
#pragma unroll
      for (int s = 0; s < 16; ++s) {
        int K = q * 32 + khi + s;
        float val = 0.f;
        if (K < KTOT) {
          int dr = K / WWIN, dc = K - dr * WWIN;
          int rr = r + dr - PADW;
          int cc = col + dc - PADW;
          if (rr >= 0 && rr < NSHEET && cc >= 0 && cc < NSHEET) {
            size_t idx = (size_t)n * N2 + (size_t)(rr * NSHEET + cc);
            val = wexc[idx] - l4c[idx] * (1.0f - mask[idx]) * inv;
          }
        }
        b[q][s] = (_Float16)val;
      }
    }
  }

  float my_netaff = (lane < 16) ? netaff[n] : 0.f;
  float cur = 0.f;                    // register copy of this neuron's resp
  float l4  = 0.f;

  for (int it = 0; it < MAXITERS; ++it) {
    __threadfence();                  // acquire: see other WGs' resp writes
    const float* rin  = respb[it & 1];
    float*       rout = respb[(it + 1) & 1];

    // stage padded resp patch (21 x 84, f16) into LDS
    for (int e = tid; e < WWIN * PATCHW; e += NTHR) {
      int dr = e / PATCHW, j = e - dr * PATCHW;
      int rr = r + dr - PADW;
      int cc = j - PADW;
      float v = 0.f;
      if (rr >= 0 && rr < NSHEET && cc >= 0 && cc < NSHEET)
        v = rin[rr * NSHEET + cc];
      patch[e] = (_Float16)v;
    }
    __syncthreads();

    // lateral = diag(A x B): A = patches, K = s + 16*(lane>=16) (matches B)
    v8f acc = {};
#pragma unroll
    for (int q = 0; q < NCH; ++q) {
      v16h a;
#pragma unroll
      for (int s = 0; s < 16; ++s) {
        int K = q * 32 + s + (lane & 16);
        _Float16 v = (_Float16)0.f;
        if (K < KTOT) {
          int dr = K / WWIN, dc = K - dr * WWIN;
          v = patch[dr * PATCHW + col + dc];
        }
        a[s] = v;
      }
      acc = __builtin_amdgcn_wmma_f32_16x16x32_f16(
          false, a, false, b[q], (short)0, acc, false, false);
    }

    // extract diagonal of the 16x16 result:
    // M<8 : VGPR M, lane M    ;  M>=8 : VGPR M-8, lane M+16
#pragma unroll
    for (int vi = 0; vi < 8; ++vi) {
      if (lane == vi)      diag[w][vi]     = acc[vi];
      if (lane == vi + 24) diag[w][vi + 8] = acc[vi];
    }
    __syncthreads();

    float change = 0.f;
    if (lane < 16) {
      float lat = diag[w][lane] * L4STR;
      float x = my_netaff + lat;
      x = x > 0.f ? x : 0.f;
      l4 = tanhf(x);
      change = fabsf(l4 - cur);
      cur = l4;
      rout[n] = l4;
    }
    // wave-wide max; non-negative floats compare correctly as uints
    for (int off = 16; off > 0; off >>= 1)
      change = fmaxf(change, __shfl_xor(change, off, 32));
    if (lane == 0)
      __hip_atomic_fetch_max(&ctl[2], __float_as_uint(change),
                             __ATOMIC_RELAXED, __HIP_MEMORY_SCOPE_AGENT);

    // grid barrier; last arriver decides convergence before releasing
    __threadfence();
    __syncthreads();
    if (tid == 0) {
      unsigned g = __hip_atomic_load(&ctl[1], __ATOMIC_RELAXED,
                                     __HIP_MEMORY_SCOPE_AGENT);
      unsigned old = __hip_atomic_fetch_add(&ctl[0], 1u, __ATOMIC_ACQ_REL,
                                            __HIP_MEMORY_SCOPE_AGENT);
      if (old == NWG - 1) {
        unsigned cb = __hip_atomic_load(&ctl[2], __ATOMIC_RELAXED,
                                        __HIP_MEMORY_SCOPE_AGENT);
        unsigned dn = (__uint_as_float(cb) < TOLV) ? 1u : 0u;
        __hip_atomic_store(&ctl[3], dn, __ATOMIC_RELAXED,
                           __HIP_MEMORY_SCOPE_AGENT);
        __hip_atomic_store(&ctl[2], 0u, __ATOMIC_RELAXED,
                           __HIP_MEMORY_SCOPE_AGENT);
        __hip_atomic_store(&ctl[0], 0u, __ATOMIC_RELAXED,
                           __HIP_MEMORY_SCOPE_AGENT);
        __hip_atomic_store(&ctl[1], g + 1u, __ATOMIC_RELEASE,
                           __HIP_MEMORY_SCOPE_AGENT);
      } else {
        while (__hip_atomic_load(&ctl[1], __ATOMIC_ACQUIRE,
                                 __HIP_MEMORY_SCOPE_AGENT) == g)
          __builtin_amdgcn_s_sleep(1);
      }
      sdone = (int)__hip_atomic_load(&ctl[3], __ATOMIC_RELAXED,
                                     __HIP_MEMORY_SCOPE_AGENT);
    }
    __syncthreads();
    if (sdone) break;
  }

  if (lane < 16) out[n] = l4;
}

// ---------------------------------------------------------------------------
extern "C" void kernel_launch(void* const* d_in, const int* in_sizes, int n_in,
                              void* d_out, int out_size, void* d_ws,
                              size_t ws_size, hipStream_t stream) {
  const float* input = (const float*)d_in[0];   // input_crop        (9216)
  const float* aff   = (const float*)d_in[1];   // afferent_weights  (4096*169)
  const float* bias  = (const float*)d_in[2];   // retinotopic_bias  (4096*169)
  const float* wexc  = (const float*)d_in[3];   // lateral_weights_exc (4096*4096)
  const float* l4c   = (const float*)d_in[4];   // l4_correlations   (4096*4096)
  const float* mask  = (const float*)d_in[5];   // masks             (4096*4096)
  const float* thr   = (const float*)d_in[6];   // l4_thresholds     (4096)
  const int*   rf    = (const int*)d_in[7];     // rf_grids          (4096*169)
  float* ws  = (float*)d_ws;
  float* out = (float*)d_out;

  k_init<<<16, 256, 0, stream>>>(ws);
  k_afferent<<<16, 256, 0, stream>>>(input, aff, bias, thr, rf, ws);
  k_midsum<<<N2, 256, 0, stream>>>(l4c, mask, ws);
  k_iterate<<<NWG, NTHR, 0, stream>>>(wexc, l4c, mask, ws, out);
}
